// FusionPipeline_24541443129392
// MI455X (gfx1250) — matrix operand order repaired and checked by reading, not compiled, not run on hardware
//
#include <hip/hip_runtime.h>
#include <hip/hip_bf16.h>

typedef __attribute__((ext_vector_type(16))) _Float16 v16h;
typedef __attribute__((ext_vector_type(8)))  float    v8f;

#define EPS_BN 1e-5f

// Block: 256 threads = 8 waves (wave32). Each wave owns a 16-row strip,
// block covers 128 rows x all 256 output features of the first GEMM.
__global__ __launch_bounds__(256) void fusion_pipeline_kernel(
    const float* __restrict__ frame, const float* __restrict__ imu,
    const float* __restrict__ ig, const float* __restrict__ ibt,
    const float* __restrict__ im, const float* __restrict__ iv,
    const float* __restrict__ fg, const float* __restrict__ fbt,
    const float* __restrict__ fm, const float* __restrict__ fv,
    const float* __restrict__ W0, const float* __restrict__ b0,
    const float* __restrict__ cg, const float* __restrict__ cbt,
    const float* __restrict__ cm, const float* __restrict__ cv,
    const float* __restrict__ W1, const float* __restrict__ b1,
    float* __restrict__ out)
{
    __shared__ float sA[512];             // per-k BN scale  (fused index)
    __shared__ float tA[512];             // per-k BN shift
    __shared__ float sFC[256];            // fc BN scale
    __shared__ float tFC[256];            // fc BN shift (b0 folded in)
    __shared__ float W1s[512];            // W1 rows [2][256]
    __shared__ __align__(16) _Float16 Bl[256 * 32];  // 16 KB: W0 chunk, n-major

    const int tid = threadIdx.x;

    // ---- stage per-feature constants (256 threads, 1 element family each) ----
    {
        // fused k = tid  -> frame BN params (concat order: frame first)
        float s0 = fg[tid] * rsqrtf(fv[tid] + EPS_BN);
        sA[tid]       = s0;
        tA[tid]       = fbt[tid] - fm[tid] * s0;
        // fused k = tid + 256 -> imu BN params
        float s1 = ig[tid] * rsqrtf(iv[tid] + EPS_BN);
        sA[tid + 256] = s1;
        tA[tid + 256] = ibt[tid] - im[tid] * s1;
        // fc BN, fold b0:  relu(s*(h + b0 - mean) + beta) = relu(s*h + t)
        float s2 = cg[tid] * rsqrtf(cv[tid] + EPS_BN);
        sFC[tid] = s2;
        tFC[tid] = s2 * (b0[tid] - cm[tid]) + cbt[tid];
        W1s[tid]       = W1[tid];
        W1s[tid + 256] = W1[tid + 256];
    }
    __syncthreads();

    const int wave = tid >> 5;
    const int lane = tid & 31;
    const int c    = lane & 15;   // column-in-tile / A-row-in-strip
    const int hi   = lane >> 4;   // half-wave select
    const int row_block = blockIdx.x * 128;
    const int arow      = row_block + wave * 16 + c;   // A row this lane feeds

    const float* __restrict__ arow_frame = frame + (size_t)arow * 256;
    const float* __restrict__ arow_imu   = imu   + (size_t)arow * 256;

    v8f acc[16];
#pragma unroll
    for (int t = 0; t < 16; ++t)
#pragma unroll
        for (int e = 0; e < 8; ++e) acc[t][e] = 0.0f;

    // ------------------- K loop: 512 in steps of 32 -------------------
    for (int ks = 0; ks < 16; ++ks) {
        // stage B chunk: thread tid owns output feature n = tid,
        // converts W0[n][ks*32 .. +31] to f16 at Bl[n*32 ..]
        {
            const float4* w4 = (const float4*)(W0 + (size_t)tid * 512 + ks * 32);
#pragma unroll
            for (int i = 0; i < 4; ++i) {
                float4 wa = w4[2 * i], wb = w4[2 * i + 1];
                union { _Float16 h[8]; uint4 u; } pk;
                pk.h[0] = (_Float16)wa.x; pk.h[1] = (_Float16)wa.y;
                pk.h[2] = (_Float16)wa.z; pk.h[3] = (_Float16)wa.w;
                pk.h[4] = (_Float16)wb.x; pk.h[5] = (_Float16)wb.y;
                pk.h[6] = (_Float16)wb.z; pk.h[7] = (_Float16)wb.w;
                *(uint4*)&Bl[tid * 32 + i * 8] = pk.u;
            }
        }
        __syncthreads();

        // A fragment built from global f32 + BN + ReLU + cvt.
        // ISA layout (16-bit A 16x32): lane holds row M=c; elems 0..7 = k0+hi*8..+7,
        // elems 8..15 = k0+16+hi*8..+7. Groups are 8-aligned -> never cross the
        // frame/imu boundary at 256.
        const int k0 = ks * 32;
        const int ka = k0 + hi * 8;
        const int kb = ka + 16;
        const float* pa = (ka < 256) ? (arow_frame + ka) : (arow_imu + ka - 256);
        const float* pb = (kb < 256) ? (arow_frame + kb) : (arow_imu + kb - 256);
        float av[8], bv[8];
        *(float4*)&av[0] = ((const float4*)pa)[0];
        *(float4*)&av[4] = ((const float4*)pa)[1];
        *(float4*)&bv[0] = ((const float4*)pb)[0];
        *(float4*)&bv[4] = ((const float4*)pb)[1];

        union { uint4 u[2]; v16h v; } af;
        {
            _Float16 ah[16];
#pragma unroll
            for (int i = 0; i < 8; ++i) {
                float xa = fmaxf(0.0f, fmaf(sA[ka + i], av[i], tA[ka + i]));
                float xb = fmaxf(0.0f, fmaf(sA[kb + i], bv[i], tA[kb + i]));
                ah[i]     = (_Float16)xa;
                ah[i + 8] = (_Float16)xb;
            }
            __builtin_memcpy(&af, ah, 32);
        }

        // 16 N-tiles: B fragment = column n = t*16+c, elems 0..7 k = hi*8..+7, elems 8..15 k = 16+hi*8..+7
#pragma unroll
        for (int t = 0; t < 16; ++t) {
            union { uint4 u[2]; v16h v; } bf;
            const _Float16* bp = &Bl[(t * 16 + c) * 32 + hi * 8];
            bf.u[0] = ((const uint4*)bp)[0];
            bf.u[1] = ((const uint4*)bp)[2];
            acc[t] = __builtin_amdgcn_wmma_f32_16x16x32_f16(
                false, af.v, false, bf.v, (short)0, acc[t], false, false);
        }
        __syncthreads();
    }

    // ------------- epilogue: fc BN + ReLU (C/D layout: row r+8*hi, col 16t+c) -------------
#pragma unroll
    for (int t = 0; t < 16; ++t) {
        const int n = t * 16 + c;
        const float s = sFC[n], tt = tFC[n];
#pragma unroll
        for (int r = 0; r < 8; ++r)
            acc[t][r] = fmaxf(0.0f, fmaf(s, acc[t][r], tt));
    }

    // ------------- second GEMM [256 -> 2] via partials + lane reduction -------------
    float p0[8], p1[8];
#pragma unroll
    for (int r = 0; r < 8; ++r) { p0[r] = 0.0f; p1[r] = 0.0f; }
#pragma unroll
    for (int t = 0; t < 16; ++t) {
        const int n = t * 16 + c;
        const float wj0 = W1s[n], wj1 = W1s[256 + n];
#pragma unroll
        for (int r = 0; r < 8; ++r) {
            p0[r] = fmaf(acc[t][r], wj0, p0[r]);
            p1[r] = fmaf(acc[t][r], wj1, p1[r]);
        }
    }
    // sum across the 16 lanes of each half-wave (xor bits 0..3; hi bit untouched)
#pragma unroll
    for (int r = 0; r < 8; ++r) {
#pragma unroll
        for (int m = 1; m < 16; m <<= 1) {
            p0[r] += __shfl_xor(p0[r], m, 32);
            p1[r] += __shfl_xor(p1[r], m, 32);
        }
    }

    // lane c = 2r + j writes out[row r+8*hi][j]; consecutive lanes -> consecutive addrs
    const int r = c >> 1, j = c & 1;
    float val = (j == 0) ? p0[r] : p1[r];
    val = fmaxf(0.0f, val + b1[j]);
    val = fminf(val, (j == 0) ? 512.0f : 384.0f);
    const int m = row_block + wave * 16 + hi * 8 + r;
    out[(size_t)m * 2 + j] = val;
}

extern "C" void kernel_launch(void* const* d_in, const int* in_sizes, int n_in,
                              void* d_out, int out_size, void* d_ws, size_t ws_size,
                              hipStream_t stream) {
    const float* frame = (const float*)d_in[0];
    const float* imu   = (const float*)d_in[1];
    const float* ig    = (const float*)d_in[2];
    const float* ib    = (const float*)d_in[3];
    const float* im    = (const float*)d_in[4];
    const float* iv    = (const float*)d_in[5];
    const float* fg    = (const float*)d_in[6];
    const float* fb    = (const float*)d_in[7];
    const float* fm    = (const float*)d_in[8];
    const float* fv    = (const float*)d_in[9];
    const float* W0    = (const float*)d_in[10];
    const float* b0    = (const float*)d_in[11];
    const float* cg    = (const float*)d_in[12];
    const float* cb    = (const float*)d_in[13];
    const float* cm    = (const float*)d_in[14];
    const float* cv    = (const float*)d_in[15];
    const float* W1    = (const float*)d_in[16];
    const float* b1    = (const float*)d_in[17];
    float* out = (float*)d_out;

    const int rows   = in_sizes[0] / 256;   // 131072
    const int blocks = rows / 128;          // 1024

    fusion_pipeline_kernel<<<blocks, 256, 0, stream>>>(
        frame, imu, ig, ib, im, iv, fg, fb, fm, fv,
        W0, b0, cg, cb, cm, cv, W1, b1, out);
}
